// HybridGNet_27771258536782
// MI455X (gfx1250) — compile-verified, matrix operand order checked by reading, not run on hardware
//
#include <hip/hip_runtime.h>

// ---------------------------------------------------------------------------
// HybridGNet forward for MI455X (gfx1250, wave32).
//  - conv / GEMM via V_WMMA_F32_16X16X32_F16 (f16 operands, f32 accum)
//  - full operand tiles staged with the Tensor Data Mover (tensor_load_to_lds,
//    s_wait_tensorcnt); partial tiles staged manually with zero padding
// ---------------------------------------------------------------------------

typedef _Float16 half_t;
typedef __attribute__((ext_vector_type(16))) _Float16 v16h;
typedef __attribute__((ext_vector_type(8)))  float    v8f;
typedef __attribute__((ext_vector_type(4)))  unsigned int v4u;
typedef __attribute__((ext_vector_type(8)))  int          v8i_;
typedef __attribute__((ext_vector_type(4)))  int          v4i_;

#define WAVE 32
#define BLK  128   // 4 waves

#if defined(__has_builtin)
#  if __has_builtin(__builtin_amdgcn_tensor_load_to_lds)
#    define HAVE_TDM 1
#  endif
#endif
#ifndef HAVE_TDM
#  define HAVE_TDM 0
#endif

union H16 { v16h v; half_t e[16]; };
union F8  { v8f  v; float  e[8];  };

// --- WMMA fragment builders from f32 LDS tiles (convert to f16 in regs) ----
// A: 16x32 row-major [m][k]; lanes 0-15 -> K {0..7,16..23}, lanes 16-31 -> +8
static __device__ __forceinline__ v16h frag_a32(const float* sm, int lane) {
  H16 u;
  const int m = lane & 15;
  const int koff = (lane >> 4) << 3;
#pragma unroll
  for (int i = 0; i < 8; ++i) {
    u.e[i]     = (half_t)sm[m * 32 + koff + i];
    u.e[i + 8] = (half_t)sm[m * 32 + koff + 16 + i];
  }
  return u.v;
}
// B: 32x16 row-major [k][n]; per-lane n fixed, K mapping mirrors A
static __device__ __forceinline__ v16h frag_b32(const float* sm, int lane) {
  H16 u;
  const int n = lane & 15;
  const int koff = (lane >> 4) << 3;
#pragma unroll
  for (int i = 0; i < 8; ++i) {
    u.e[i]     = (half_t)sm[(koff + i) * 16 + n];
    u.e[i + 8] = (half_t)sm[(koff + 16 + i) * 16 + n];
  }
  return u.v;
}

// --- Tensor Data Mover: 2D strided f32 tile -> contiguous LDS --------------
// D# per CDNA5 ISA. tensor_dims set huge (tiles are issued
// only when fully in-bounds, so the OOB path is never exercised).
#if HAVE_TDM
static __device__ __forceinline__ void tdm_load_tile_f32(const void* lds_dst,
                                                         const float* gsrc,
                                                         unsigned tile_d0,
                                                         unsigned tile_d1,
                                                         unsigned long long stride_elems) {
  const unsigned long long ga = (unsigned long long)(uintptr_t)gsrc;
  const unsigned lds_off = (unsigned)(uintptr_t)lds_dst;  // LDS aperture: low 32 bits = offset
  v4u g0;
  g0[0] = 1u;                                        // count=1, user descriptor
  g0[1] = lds_off;                                   // lds_addr (bytes)
  g0[2] = (unsigned)(ga & 0xffffffffu);              // global_addr[31:0]
  g0[3] = (unsigned)((ga >> 32) & 0x01ffffffu) | (2u << 30);  // addr[56:32] | type=2
  v8i_ g1;
  g1[0] = (int)(2u << 16);                           // data_size = 4 bytes
  g1[1] = 0;                                         // tensor_dim0[15:0]=0 (dim0 = 2^30)
  g1[2] = 0x4000;                                    // tensor_dim0[31:16]; tensor_dim1 lo = 0
  g1[3] = (int)(0x4000u | (tile_d0 << 16));          // tensor_dim1[31:16]=2^30 hi; tile_dim0
  g1[4] = (int)(tile_d1 & 0xffffu);                  // tile_dim1; tile_dim2 = 0
  g1[5] = (int)(stride_elems & 0xffffffffull);       // tensor_dim0_stride[31:0]
  g1[6] = (int)((stride_elems >> 32) & 0xffffull);   // stride[47:32]; dim1_stride lo = 0
  g1[7] = 0;
  v4i_ z4 = {0, 0, 0, 0};
  v8i_ z8 = {0, 0, 0, 0, 0, 0, 0, 0};
  __builtin_amdgcn_tensor_load_to_lds(g0, g1, z4, z4, z8, 0);
}
#endif

// ---------------------------------------------------------------------------
// Implicit-GEMM conv (ksz 1 or 3, SAME). Block = 4 waves, tile 16(Cout) x
// 64(pixels): shared weight tile (TDM when full), per-wave im2col tile.
// Fused residual add + ReLU epilogue, f16 activations.
// ---------------------------------------------------------------------------
__global__ __launch_bounds__(BLK)
void conv_wmma_kernel(const void* __restrict__ inp, int in_is_f32,
                      const float* __restrict__ Wt,
                      const half_t* __restrict__ resid,
                      half_t* __restrict__ outp,
                      int NB, int Cin, int H, int Wd, int Cout, int ksz, int do_relu) {
  __shared__ float smA[16 * 32];        // weights [oc][k]
  __shared__ float smB[4][32 * 16];     // per-wave patches [k][pix]
  const int lane = threadIdx.x & (WAVE - 1);
  const int wv   = threadIdx.x >> 5;
  const int P   = NB * H * Wd;
  const int n0  = (blockIdx.x * 4 + wv) * 16;
  const int oc0 = blockIdx.y * 16;
  const int KK  = Cin * ksz * ksz;
  const int kh  = ksz >> 1;
  const int k2  = ksz * ksz;
  const float*  in32 = (const float*)inp;
  const half_t* in16 = (const half_t*)inp;
  v8f c = {};

  for (int k0 = 0; k0 < KK; k0 += 32) {
    if (k0 + 32 < KK)
      __builtin_prefetch(&Wt[(size_t)oc0 * KK + k0 + 32], 0, 1);  // global_prefetch_b8
    // ---- stage A (shared weight tile) ----
    const bool fullA = (k0 + 32 <= KK) && (oc0 + 16 <= Cout);
#if HAVE_TDM
    if (fullA) {
      if (wv == 0)
        tdm_load_tile_f32(smA, Wt + (size_t)oc0 * KK + k0, 32u, 16u,
                          (unsigned long long)KK);
    } else
#endif
    {
      for (int idx = threadIdx.x; idx < 512; idx += BLK) {
        const int m = idx >> 5, kk = idx & 31;
        const int gm = oc0 + m, gk = k0 + kk;
        smA[idx] = (gm < Cout && gk < KK) ? Wt[(size_t)gm * KK + gk] : 0.f;
      }
    }
    // ---- stage B (per-wave im2col patch tile) ----
    for (int idx = lane; idx < 512; idx += WAVE) {
      const int kk = idx >> 4, n = idx & 15;
      const int gk = k0 + kk, p = n0 + n;
      float v = 0.f;
      if (gk < KK && p < P) {
        const int ci = gk / k2;
        const int rc = gk - ci * k2;
        const int dy = rc / ksz - kh;
        const int dx = rc - (rc / ksz) * ksz - kh;
        const int b   = p / (H * Wd);
        const int rem = p - b * H * Wd;
        const int py  = rem / Wd;
        const int px  = rem - py * Wd;
        const int y = py + dy, x = px + dx;
        if (y >= 0 && y < H && x >= 0 && x < Wd) {
          const size_t off = (((size_t)b * Cin + ci) * H + y) * Wd + x;
          v = in_is_f32 ? in32[off] : (float)in16[off];
        }
      }
      smB[wv][kk * 16 + n] = v;
    }
#if HAVE_TDM
    __builtin_amdgcn_s_wait_tensorcnt(0);
#endif
    __syncthreads();
    const v16h a = frag_a32(smA, lane);
    const v16h b = frag_b32(smB[wv], lane);
    c = __builtin_amdgcn_wmma_f32_16x16x32_f16(false, a, false, b, (short)0, c,
                                               false, false);
    __syncthreads();
  }

  F8 cu; cu.v = c;
#pragma unroll
  for (int r = 0; r < 8; ++r) {
    const int m = r + ((lane >> 4) << 3);
    const int n = lane & 15;
    const int oc = oc0 + m, p = n0 + n;
    if (oc < Cout && p < P) {
      const int b   = p / (H * Wd);
      const int rem = p - b * H * Wd;
      const int py  = rem / Wd;
      const int px  = rem - py * Wd;
      const size_t off = (((size_t)b * Cout + oc) * H + py) * Wd + px;
      float v = cu.e[r];
      if (resid) v += (float)resid[off];
      if (do_relu) v = fmaxf(v, 0.f);
      outp[off] = (half_t)v;
    }
  }
}

// ---------------------------------------------------------------------------
// Generic WMMA GEMM: C[M,N] (f32) (+)= A[M,K] * B[K,N] (+bias, ReLU).
// Block = 4 waves, tile 64(M) x 16(N): shared B tile (TDM by wave 0 when
// full), per-wave A tile (own TDM when full).
// ---------------------------------------------------------------------------
__global__ __launch_bounds__(BLK)
void gemm_wmma_kernel(const float* __restrict__ A, int lda,
                      const float* __restrict__ Bm, int ldb,
                      float* __restrict__ C, int ldc,
                      int M, int N, int K,
                      const float* __restrict__ bias, int do_relu, int accum) {
  __shared__ float smA[4][16 * 32];   // per-wave [m][k]
  __shared__ float smB[32 * 16];      // shared  [k][n]
  const int lane = threadIdx.x & (WAVE - 1);
  const int wv   = threadIdx.x >> 5;
  const int m0 = (blockIdx.x * 4 + wv) * 16;
  const int n0 = blockIdx.y * 16;
  v8f c = {};

  for (int k0 = 0; k0 < K; k0 += 32) {
    if (k0 + 32 < K)
      __builtin_prefetch(&Bm[(size_t)(k0 + 32) * ldb + n0], 0, 1);
    // ---- stage B (shared) ----
    const bool fullB = (k0 + 32 <= K) && (n0 + 16 <= N);
#if HAVE_TDM
    if (fullB) {
      if (wv == 0)
        tdm_load_tile_f32(smB, Bm + (size_t)k0 * ldb + n0, 16u, 32u,
                          (unsigned long long)ldb);
    } else
#endif
    {
      for (int idx = threadIdx.x; idx < 512; idx += BLK) {
        const int kk = idx >> 4, n = idx & 15;
        const int gk = k0 + kk, gn = n0 + n;
        smB[kk * 16 + n] = (gk < K && gn < N) ? Bm[(size_t)gk * ldb + gn] : 0.f;
      }
    }
    // ---- stage A (per wave) ----
    const bool fullA = (k0 + 32 <= K) && (m0 + 16 <= M);
#if HAVE_TDM
    if (fullA) {
      tdm_load_tile_f32(smA[wv], A + (size_t)m0 * lda + k0, 32u, 16u,
                        (unsigned long long)lda);
    } else
#endif
    {
      for (int idx = lane; idx < 512; idx += WAVE) {
        const int m = idx >> 5, kk = idx & 31;
        const int gm = m0 + m, gk = k0 + kk;
        smA[wv][idx] = (gm < M && gk < K) ? A[(size_t)gm * lda + gk] : 0.f;
      }
    }
#if HAVE_TDM
    __builtin_amdgcn_s_wait_tensorcnt(0);
#endif
    __syncthreads();
    const v16h a = frag_a32(smA[wv], lane);
    const v16h b = frag_b32(smB, lane);
    c = __builtin_amdgcn_wmma_f32_16x16x32_f16(false, a, false, b, (short)0, c,
                                               false, false);
    __syncthreads();
  }

  F8 cu; cu.v = c;
#pragma unroll
  for (int r = 0; r < 8; ++r) {
    const int m = r + ((lane >> 4) << 3);
    const int n = lane & 15;
    const int gm = m0 + m, gn = n0 + n;
    if (gm < M && gn < N) {
      float v = cu.e[r];
      if (accum) v += C[(size_t)gm * ldc + gn];
      if (bias)  v += bias[gn];
      if (do_relu) v = fmaxf(v, 0.f);
      C[(size_t)gm * ldc + gn] = v;
    }
  }
}

// ---------------------------------------------------------------------------
// Small elementwise helpers
// ---------------------------------------------------------------------------
__global__ void maxpool2_kernel(const half_t* __restrict__ in, half_t* __restrict__ out,
                                int NB, int C, int H, int W) {
  const int Ho = H >> 1, Wo = W >> 1;
  const int total = NB * C * Ho * Wo;
  const int i = blockIdx.x * blockDim.x + threadIdx.x;
  if (i >= total) return;
  int t = i;
  const int x = t % Wo; t /= Wo;
  const int y = t % Ho; t /= Ho;
  const int ch = t % C;
  const int b = t / C;
  const half_t* base = in + ((size_t)b * C + ch) * H * W;
  const float v00 = (float)base[(2 * y) * W + 2 * x];
  const float v01 = (float)base[(2 * y) * W + 2 * x + 1];
  const float v10 = (float)base[(2 * y + 1) * W + 2 * x];
  const float v11 = (float)base[(2 * y + 1) * W + 2 * x + 1];
  out[(((size_t)b * C + ch) * Ho + y) * Wo + x] =
      (half_t)fmaxf(fmaxf(v00, v01), fmaxf(v10, v11));
}

__global__ void half2float_kernel(const half_t* __restrict__ in, float* __restrict__ out, int n) {
  const int i = blockIdx.x * blockDim.x + threadIdx.x;
  if (i < n) out[i] = (float)in[i];
}

__global__ void zero_kernel(float* __restrict__ p, int n) {
  const int i = blockIdx.x * blockDim.x + threadIdx.x;
  if (i < n) p[i] = 0.f;
}

__global__ void upsample_kernel(const float* __restrict__ g, float* __restrict__ o,
                                const int* __restrict__ idx, const float* __restrict__ w,
                                int Nf, int Nc, int F) {
  const int total = 2 * Nf * F;
  const int i = blockIdx.x * blockDim.x + threadIdx.x;
  if (i >= total) return;
  int t = i;
  const int f = t % F; t /= F;
  const int n = t % Nf;
  const int b = t / Nf;
  const float* gb = g + (size_t)b * Nc * F;
  float acc = 0.f;
#pragma unroll
  for (int j = 0; j < 3; ++j)
    acc += gb[(size_t)idx[n * 3 + j] * F + f] * w[n * 3 + j];
  o[i] = acc;
}

__global__ void prop_kernel(const float* __restrict__ t, float* __restrict__ o,
                            const int* __restrict__ row, const int* __restrict__ col,
                            const float* __restrict__ norm, int E, int Nn, int F) {
  const int i = blockIdx.x * blockDim.x + threadIdx.x;
  if (i >= E * F) return;
  const int e = i / F, f = i - e * F;
  const float nv = norm[e];
  const int r = row[e], cc = col[e];
#pragma unroll
  for (int b = 0; b < 2; ++b)
    atomicAdd(&o[((size_t)b * Nn + r) * F + f],
              t[((size_t)b * Nn + cc) * F + f] * nv);
}

__global__ void combine_kernel(float* __restrict__ p, const float* __restrict__ t0, int n) {
  const int i = blockIdx.x * blockDim.x + threadIdx.x;
  if (i < n) p[i] = 2.f * p[i] - t0[i];
}

__global__ void bias_relu_kernel(float* __restrict__ o, const float* __restrict__ bias,
                                 int n, int F) {
  const int i = blockIdx.x * blockDim.x + threadIdx.x;
  if (i < n) o[i] = fmaxf(o[i] + bias[i % F], 0.f);
}

// ---------------------------------------------------------------------------
// Host orchestration
// ---------------------------------------------------------------------------
static inline int ceil_div(int a, int b) { return (a + b - 1) / b; }

extern "C" void kernel_launch(void* const* d_in, const int* in_sizes, int n_in,
                              void* d_out, int out_size, void* d_ws, size_t ws_size,
                              hipStream_t stream) {
  (void)in_sizes; (void)n_in; (void)out_size; (void)ws_size;

  const float* x = (const float*)d_in[0];
  const float *w1[6], *w2[6], *wsk[6];
  int wi = 1;
  for (int b = 0; b < 6; ++b) {
    w1[b] = (const float*)d_in[wi++];
    w2[b] = (const float*)d_in[wi++];
    wsk[b] = (b < 5) ? (const float*)d_in[wi++] : nullptr;
  }
  const float* fc_w = (const float*)d_in[18];
  const float* fc_b = (const float*)d_in[19];
  const float* dl_w = (const float*)d_in[20];
  const float* dl_b = (const float*)d_in[21];
  const float* chW[5];  for (int i = 0; i < 5; ++i) chW[i] = (const float*)d_in[22 + i];
  const float* chB[4];  for (int i = 0; i < 4; ++i) chB[i] = (const float*)d_in[27 + i];
  // rows 31..34, cols 35..38, norms 39..42, up_idx 43..46, up_w 47..50 (lvl 0..3)

  const size_t GB = 33554432;  // 32 MiB slabs
  char* R = (char*)d_ws;
  half_t* bA = (half_t*)(R);
  half_t* bB = (half_t*)(R + GB);
  half_t* bC = (half_t*)(R + 2 * GB);
  float* OUTB = (float*)(R + 3 * GB);
  float* Gr[3] = { (float*)(R), (float*)(R + GB), (float*)(R + 2 * GB) };
  float* HF = (float*)(R + 4 * GB);
  float* ZB = (float*)(R + 4 * GB + (1 << 20));

  const int ENC[7] = {1, 8, 16, 32, 64, 128, 128};

  auto conv = [&](const void* in, int in_f32, const float* Wt, const half_t* resid,
                  half_t* out, int Cin, int HH, int Cout, int ksz, int relu) {
    dim3 g(ceil_div(2 * HH * HH, 64), ceil_div(Cout, 16));
    conv_wmma_kernel<<<g, BLK, 0, stream>>>(in, in_f32, Wt, resid, out,
                                            2, Cin, HH, HH, Cout, ksz, relu);
  };
  auto gemm = [&](const float* A, int lda, const float* B, int ldb, float* C, int ldc,
                  int M, int N, int K, const float* bias, int relu, int accum) {
    dim3 g(ceil_div(M, 64), ceil_div(N, 16));
    gemm_wmma_kernel<<<g, BLK, 0, stream>>>(A, lda, B, ldb, C, ldc, M, N, K, bias, relu, accum);
  };
  auto zero = [&](float* p, int n) {
    zero_kernel<<<ceil_div(n, 256), 256, 0, stream>>>(p, n);
  };

  // ---- CNN encoder ----
  conv(x, 1, wsk[0], nullptr, bA, 1, 1024, 8, 1, 0);
  conv(x, 1, w1[0], nullptr, bB, 1, 1024, 8, 3, 1);
  conv(bB, 0, w2[0], bA, bC, 8, 1024, 8, 3, 1);
  maxpool2_kernel<<<ceil_div(2 * 8 * 512 * 512, 256), 256, 0, stream>>>(bC, bA, 2, 8, 1024, 1024);

  half_t* cur = bA; half_t* f1 = bB; half_t* f2 = bC;
  int HH = 512;
  for (int b = 1; b <= 4; ++b) {
    const int ci = ENC[b], co = ENC[b + 1];
    conv(cur, 0, wsk[b], nullptr, f1, ci, HH, co, 1, 0);
    conv(cur, 0, w1[b], nullptr, f2, ci, HH, co, 3, 1);
    conv(f2, 0, w2[b], f1, cur, co, HH, co, 3, 1);
    maxpool2_kernel<<<ceil_div(2 * co * (HH / 2) * (HH / 2), 256), 256, 0, stream>>>(
        cur, f1, 2, co, HH, HH);
    half_t* nin = f1; f1 = f2; f2 = cur; cur = nin;
    HH >>= 1;
  }
  conv(cur, 0, w1[5], nullptr, f1, 128, 32, 128, 3, 1);
  conv(f1, 0, w2[5], cur, f2, 128, 32, 128, 3, 1);
  half_t* hfin = f2;

  // ---- latent + decoder linear ----
  half2float_kernel<<<ceil_div(262144, 256), 256, 0, stream>>>(hfin, HF, 262144);
  gemm(HF, 131072, fc_w, 64, ZB, 64, 2, 64, 131072, fc_b, 0, 0);
  gemm(ZB, 64, dl_w, 16384, OUTB, 16384, 2, 16384, 64, dl_b, 1, 0);

  // ---- graph decoder ----
  const int SIZESa[5] = {131072, 32768, 8192, 2048, 512};
  const int FI[5] = {32, 32, 32, 32, 16};
  const int FO[5] = {32, 32, 32, 16, 3};

  auto cheb = [&](float* xin, float* s1, float* s2, const float* Wk, const float* bias,
                  int lvl, int Nn, int Fin, int Fout, float* outp) {
    const int* rowp = (const int*)d_in[31 + lvl];
    const int* colp = (const int*)d_in[35 + lvl];
    const float* nrm = (const float*)d_in[39 + lvl];
    const int E = 6 * Nn, M = 2 * Nn, NT = 2 * Nn * Fin;
    gemm(xin, Fin, Wk, Fout, outp, Fout, M, Fout, Fin, nullptr, 0, 0);
    zero(s1, NT);
    prop_kernel<<<ceil_div(E * Fin, 256), 256, 0, stream>>>(xin, s1, rowp, colp, nrm, E, Nn, Fin);
    gemm(s1, Fin, Wk + 1 * Fin * Fout, Fout, outp, Fout, M, Fout, Fin, nullptr, 0, 1);
    float* t0 = xin; float* t1 = s1; float* sp = s2;
    for (int k = 2; k < 6; ++k) {
      zero(sp, NT);
      prop_kernel<<<ceil_div(E * Fin, 256), 256, 0, stream>>>(t1, sp, rowp, colp, nrm, E, Nn, Fin);
      combine_kernel<<<ceil_div(NT, 256), 256, 0, stream>>>(sp, t0, NT);
      gemm(sp, Fin, Wk + k * Fin * Fout, Fout, outp, Fout, M, Fout, Fin, nullptr, 0, 1);
      float* dead = t0; t0 = t1; t1 = sp; sp = dead;
    }
    if (bias)
      bias_relu_kernel<<<ceil_div(M * Fout, 256), 256, 0, stream>>>(outp, bias, M * Fout, Fout);
  };

  for (int i = 0; i < 4; ++i) {
    const int lvl = 3 - i;
    const int Nn = SIZESa[lvl], Nc = SIZESa[lvl + 1];
    const int Fin = FI[i], Fout = FO[i];
    upsample_kernel<<<ceil_div(2 * Nn * Fin, 256), 256, 0, stream>>>(
        OUTB, Gr[0], (const int*)d_in[43 + lvl], (const float*)d_in[47 + lvl], Nn, Nc, Fin);
    cheb(Gr[0], Gr[1], Gr[2], chW[i], chB[i], lvl, Nn, Fin, Fout, OUTB);
  }
  cheb(OUTB, Gr[0], Gr[1], chW[4], nullptr, 0, 131072, FI[4], FO[4], (float*)d_out);
}
